// GCN_81449759801465
// MI455X (gfx1250) — matrix operand order repaired and checked by reading, not compiled, not run on hardware
//
#include <hip/hip_runtime.h>
#include <hip/hip_bf16.h>
#include <math.h>

typedef __attribute__((ext_vector_type(16))) _Float16 v16h;
typedef __attribute__((ext_vector_type(8)))  float    v8f;

#define HIDDEN   64
#define N_LABELS 40
#define W2FRAG_ELEMS (3 * 2 * 32 * 16)   // [nt][kc][lane][16] f16 fragments

// ---------------------------------------------------------------------------
// Layer-1 feature SpMM: h[r,:] += v * W1[c,:]   (wave per nonzero, 64 dims)
// W1 is 512KB -> resident in L2; h (25.6MB) -> atomics resolve in L2.
// ---------------------------------------------------------------------------
__global__ void feat_spmm(const long long* __restrict__ fidx,
                          const float* __restrict__ fval,
                          const float* __restrict__ W1,
                          float* __restrict__ h, int nnz) {
  int wid  = (int)((blockIdx.x * blockDim.x + threadIdx.x) >> 5);
  int lane = threadIdx.x & 31;
  if (wid >= nnz) return;                       // wave-uniform exit
  long long r = fidx[wid];
  long long c = fidx[(long long)nnz + wid];
  float v = fval[wid];
  const float* wrow = W1 + c * HIDDEN;
  float*       hrow = h  + r * HIDDEN;
  atomicAdd(&hrow[lane],      v * wrow[lane]);
  atomicAdd(&hrow[lane + 32], v * wrow[lane + 32]);
}

__global__ void add_b1(float* __restrict__ h, const float* __restrict__ b1, int n) {
  int i = blockIdx.x * blockDim.x + threadIdx.x;
  if (i < n) h[i] += b1[i & (HIDDEN - 1)];
}

// ---------------------------------------------------------------------------
// Adjacency SpMM (64-wide): h2[r,:] += w * h[c,:]   (wave per edge)
// ---------------------------------------------------------------------------
__global__ void edge_spmm_h(const long long* __restrict__ eidx,
                            const float* __restrict__ ew,
                            const float* __restrict__ h,
                            float* __restrict__ h2, int ne) {
  int wid  = (int)((blockIdx.x * blockDim.x + threadIdx.x) >> 5);
  int lane = threadIdx.x & 31;
  if (wid >= ne) return;
  long long r = eidx[wid];
  long long c = eidx[(long long)ne + wid];
  float w = ew[wid];
  const float* src = h  + c * HIDDEN;
  float*       dst = h2 + r * HIDDEN;
  atomicAdd(&dst[lane],      w * src[lane]);
  atomicAdd(&dst[lane + 32], w * src[lane + 32]);
}

// ---------------------------------------------------------------------------
// One-shot W2 -> f16 B-fragment pack, in exact WMMA per-lane order:
//   w2f[((nt*2 + kc)*32 + lane)*16 + j]
//     = f16( W2[ (kc*32 + (j>=8)*16 + (lane>=16)*8 + (j&7)) * 40 + clamp(nt*16 + (lane&15)) ] )
// Out-of-range cols (tile 2, lanes 8..15) clamp to col 39: duplicate-but-valid
// data whose accumulator columns are never stored.
// ---------------------------------------------------------------------------
__global__ void pack_w2(const float* __restrict__ W2, _Float16* __restrict__ w2f) {
  int idx = blockIdx.x * blockDim.x + threadIdx.x;
  if (idx >= W2FRAG_ELEMS) return;
  int j    = idx & 15;
  int lane = (idx >> 4) & 31;
  int kc   = (idx >> 9) & 1;
  int nt   = idx >> 10;
  int K    = kc * 32 + (j >> 3) * 16 + (lane >> 4) * 8 + (j & 7);
  int col  = nt * 16 + (lane & 15);
  if (col >= N_LABELS) col = N_LABELS - 1;
  w2f[idx] = (_Float16)W2[K * N_LABELS + col];
}

// ---------------------------------------------------------------------------
// Dense layer 2 via WMMA:  z[100000,40] = relu(h2) @ W2 + b2
// One wave computes a 16-row M-tile; K=64 as two 16x16x32 f16 WMMAs per
// N-tile; N=40 -> tiles 0,1 fully valid, tile 2 stores only cols 32..39.
// B fragments are pre-packed (pack_w2) -> contiguous v16h loads per wave.
//
// A layout (16-bit A 16x32, ISA 7.12.2): lane L holds row M=L&15;
//   low-half lanes: K = j (j<8), 16+j-8 (j>=8); high-half lanes: +8.
// D layout (f32 16x16): lane -> N=L&15, vgpr r -> M = r + 8*(L>=16).
// ---------------------------------------------------------------------------
__global__ void gemm_wmma(const float* __restrict__ h2,
                          const _Float16* __restrict__ w2f,
                          const float* __restrict__ b2,
                          float* __restrict__ z, int mtiles, int nrows) {
  int wave = (int)((blockIdx.x * blockDim.x + threadIdx.x) >> 5);
  int lane = threadIdx.x & 31;
  if (wave >= mtiles) return;                   // wave-uniform: EXEC stays full
  int  m0   = wave * 16;
  bool full = (m0 + 16 <= nrows);               // wave-uniform fast-path flag
  int  mrow = m0 + (lane & 15);
  int  mclamp = mrow < nrows ? mrow : (nrows - 1);
  int  hi   = lane >> 4;                        // 0 = lanes 0-15, 1 = 16-31
  int  ncol = lane & 15;
  bool n2valid = ncol < 8;                      // tile-2 store validity

  // Load A tile with fused ReLU + f32->f16 convert, for both K chunks.
  v16h a[2];
  const float* arow = h2 + (long long)mclamp * HIDDEN;
  #pragma unroll
  for (int kc = 0; kc < 2; ++kc) {
    int base = kc * 32 + hi * 8;
    #pragma unroll
    for (int j = 0; j < 8; ++j) {
      float x0 = arow[base + j];
      float x1 = arow[base + 16 + j];
      a[kc][j]     = (_Float16)(x0 > 0.f ? x0 : 0.f);
      a[kc][j + 8] = (_Float16)(x1 > 0.f ? x1 : 0.f);
    }
  }

  // B fragments: contiguous 32B (v16h) loads from the packed buffer.
  const v16h* bf = (const v16h*)w2f;
  v8f acc0 = {}, acc1 = {}, acc2 = {};
  #pragma unroll
  for (int kc = 0; kc < 2; ++kc) {
    v16h b0  = bf[(0 * 2 + kc) * 32 + lane];
    v16h b1  = bf[(1 * 2 + kc) * 32 + lane];
    v16h b2v = bf[(2 * 2 + kc) * 32 + lane];
    // (neg_a, A, neg_b, B, c_mod, C, reuse_a, reuse_b)
    acc0 = __builtin_amdgcn_wmma_f32_16x16x32_f16(false, a[kc], false, b0,
                                                  (short)0, acc0, false, false);
    acc1 = __builtin_amdgcn_wmma_f32_16x16x32_f16(false, a[kc], false, b1,
                                                  (short)0, acc1, false, false);
    acc2 = __builtin_amdgcn_wmma_f32_16x16x32_f16(false, a[kc], false, b2v,
                                                  (short)0, acc2, false, false);
  }

  int rbase = m0 + hi * 8;                      // first of this lane's 8 rows
  float bias0 = b2[ncol];
  float bias1 = b2[16 + ncol];
  float bias2 = b2[32 + (ncol & 7)];            // branchless in-bounds clamp

  if (full) {
    // Fast path: one base pointer, stores use compile-time immediate offsets.
    float* p = z + (long long)rbase * N_LABELS + ncol;
    #pragma unroll
    for (int r = 0; r < 8; ++r) {
      p[r * N_LABELS]      = acc0[r] + bias0;
      p[r * N_LABELS + 16] = acc1[r] + bias1;
    }
    if (n2valid) {
      #pragma unroll
      for (int r = 0; r < 8; ++r)
        p[r * N_LABELS + 32] = acc2[r] + bias2;
    }
  } else {
    // Tail path (only for a partial final tile).
    #pragma unroll
    for (int r = 0; r < 8; ++r) {
      int row = rbase + r;
      if (row < nrows) {
        float* p = z + (long long)row * N_LABELS + ncol;
        p[0]  = acc0[r] + bias0;
        p[16] = acc1[r] + bias1;
        if (n2valid) p[32] = acc2[r] + bias2;
      }
    }
  }
}

// ---------------------------------------------------------------------------
// Adjacency SpMM (40-wide) into pre-zeroed d_out: out[r,:] += w * z[c,:]
// ---------------------------------------------------------------------------
__global__ void edge_spmm_z(const long long* __restrict__ eidx,
                            const float* __restrict__ ew,
                            const float* __restrict__ z,
                            float* __restrict__ out, int ne) {
  int wid  = (int)((blockIdx.x * blockDim.x + threadIdx.x) >> 5);
  int lane = threadIdx.x & 31;
  if (wid >= ne) return;
  long long r = eidx[wid];
  long long c = eidx[(long long)ne + wid];
  float w = ew[wid];
  const float* src = z   + c * N_LABELS;
  float*       dst = out + r * N_LABELS;
  atomicAdd(&dst[lane], w * src[lane]);                         // cols 0..31
  if (lane < 8) atomicAdd(&dst[lane + 32], w * src[lane + 32]); // cols 32..39
}

// ---------------------------------------------------------------------------
// In-place row log-softmax over 40 cols (wave per row, shuffle reductions)
// ---------------------------------------------------------------------------
__global__ void logsoftmax_rows(float* __restrict__ out, int nrows) {
  int row  = (int)((blockIdx.x * blockDim.x + threadIdx.x) >> 5);
  int lane = threadIdx.x & 31;
  if (row >= nrows) return;
  float* p = out + (long long)row * N_LABELS;
  float v0 = p[lane];
  float v1 = (lane < 8) ? p[lane + 32] : -1e30f;
  float m = fmaxf(v0, v1);
  #pragma unroll
  for (int off = 16; off > 0; off >>= 1)
    m = fmaxf(m, __shfl_xor(m, off, 32));
  float s = __expf(v0 - m) + ((lane < 8) ? __expf(v1 - m) : 0.f);
  #pragma unroll
  for (int off = 16; off > 0; off >>= 1)
    s += __shfl_xor(s, off, 32);
  float lse = m + __logf(s);
  p[lane] = v0 - lse;
  if (lane < 8) p[lane + 32] = v1 - lse;
}

// ---------------------------------------------------------------------------
extern "C" void kernel_launch(void* const* d_in, const int* in_sizes, int n_in,
                              void* d_out, int out_size, void* d_ws, size_t ws_size,
                              hipStream_t stream) {
  const long long* fidx = (const long long*)d_in[0];   // [2, NNZ] int64
  const float*     fval = (const float*)    d_in[1];   // [NNZ]
  const long long* eidx = (const long long*)d_in[2];   // [2, E]  int64
  const float*     ew   = (const float*)    d_in[3];   // [E]
  const float*     W1   = (const float*)    d_in[4];   // [2048, 64]
  const float*     b1   = (const float*)    d_in[5];   // [64]
  const float*     W2   = (const float*)    d_in[6];   // [64, 40]
  const float*     b2   = (const float*)    d_in[7];   // [40]
  float*           out  = (float*)d_out;               // [N, 40]

  int nnz   = in_sizes[1];
  int ne    = in_sizes[3];
  int nodes = out_size / N_LABELS;

  // Workspace: h [N*64] | h2 [N*64] | w2frag [3072 f16]; z aliases h.
  float*     h    = (float*)d_ws;
  float*     h2   = h + (size_t)nodes * HIDDEN;
  _Float16*  w2f  = (_Float16*)(h2 + (size_t)nodes * HIDDEN);
  float*     z    = h;

  hipMemsetAsync(h,   0, (size_t)nodes * HIDDEN * sizeof(float) * 2, stream);
  hipMemsetAsync(out, 0, (size_t)out_size * sizeof(float), stream);

  const int T = 256;  // 8 waves per block

  {  // pack W2 into WMMA B fragments (independent of the SpMM chain)
    pack_w2<<<(W2FRAG_ELEMS + T - 1) / T, T, 0, stream>>>(W2, w2f);
  }
  {  // layer-1 SpMM: one wave per nnz
    long long threads = (long long)nnz * 32;
    int blocks = (int)((threads + T - 1) / T);
    feat_spmm<<<blocks, T, 0, stream>>>(fidx, fval, W1, h, nnz);
  }
  {  // + b1
    int n = nodes * HIDDEN;
    add_b1<<<(n + T - 1) / T, T, 0, stream>>>(h, b1, n);
  }
  {  // adjacency propagate (64-wide)
    long long threads = (long long)ne * 32;
    int blocks = (int)((threads + T - 1) / T);
    edge_spmm_h<<<blocks, T, 0, stream>>>(eidx, ew, h, h2, ne);
  }
  {  // dense layer 2 via WMMA (ReLU + bias fused)
    int mtiles = (nodes + 15) / 16;
    long long threads = (long long)mtiles * 32;
    int blocks = (int)((threads + T - 1) / T);
    gemm_wmma<<<blocks, T, 0, stream>>>(h2, w2f, b2, z, mtiles, nodes);
  }
  {  // adjacency propagate (40-wide) into d_out
    long long threads = (long long)ne * 32;
    int blocks = (int)((threads + T - 1) / T);
    edge_spmm_z<<<blocks, T, 0, stream>>>(eidx, ew, z, out, ne);
  }
  {  // log-softmax rows in place
    long long threads = (long long)nodes * 32;
    int blocks = (int)((threads + T - 1) / T);
    logsoftmax_rows<<<blocks, T, 0, stream>>>(out, nodes);
  }
}
